// MultilayerGRU_68856915690256
// MI455X (gfx1250) — compile-verified, matrix operand order checked by reading, not run on hardware
//
#include <hip/hip_runtime.h>
#include <hip/hip_bf16.h>

// Multilayer GRU on MI455X (gfx1250, wave32).
// - All GEMMs via V_WMMA_F32_16X16X32_F16 (f16 operands, f32 accumulate).
// - Weights + all GEMM "A" activations kept f16 in d_ws; no cvt in hot loops.
// - A slabs staged into LDS with GLOBAL_LOAD_ASYNC_TO_LDS_B128 (ASYNCcnt).
// - The 256-step recurrent scan of each layer runs as ONE persistent kernel
//   (32 blocks) with an atomic grid barrier between the z/r and g phases.

typedef __attribute__((ext_vector_type(16))) _Float16 v16h;
typedef __attribute__((ext_vector_type(8)))  _Float16 v8h;
typedef __attribute__((ext_vector_type(8)))  float    v8f;

#define BB 64
#define SS 256
#define II 512
#define HH 1024
#define OO 512

// ---------- WMMA helpers (CDNA5 VGPR layouts, wave32) ----------

__device__ __forceinline__ v8f wmma_f16(v16h a, v16h b, v8f c) {
  // (neg_a, A, neg_b, B, c_mod, C, reuse_a, reuse_b)
  return __builtin_amdgcn_wmma_f32_16x16x32_f16(false, a, false, b, (short)0, c,
                                                false, false);
}

// 16x32 f16 operand tile load (A layout; also valid as B when B[k][n]=W[n][k],
// W row-major): row = lane&15; lanes 0-15 take K{0..7,16..23},
// lanes 16-31 take K{8..15,24..31}.
__device__ __forceinline__ v16h load_tile_f16(const _Float16* __restrict__ p, int ld) {
  const int lane = threadIdx.x & 31;
  const _Float16* row = p + (size_t)(lane & 15) * ld + (lane >> 4) * 8;
  v8h lo = *(const v8h*)(row);
  v8h hi = *(const v8h*)(row + 16);
  v16h out;
#pragma unroll
  for (int i = 0; i < 8; ++i) { out[i] = lo[i]; out[8 + i] = hi[i]; }
  return out;
}

// Same tile, but from an LDS-resident A slab (16 rows x ld halves).
__device__ __forceinline__ v16h load_tile_lds(const _Float16* s, int ld, int k) {
  const int lane = threadIdx.x & 31;
  const _Float16* row = s + (size_t)(lane & 15) * ld + k + (lane >> 4) * 8;
  v8h lo = *(const v8h*)(row);
  v8h hi = *(const v8h*)(row + 16);
  v16h out;
#pragma unroll
  for (int i = 0; i < 8; ++i) { out[i] = lo[i]; out[8 + i] = hi[i]; }
  return out;
}

// Cooperative async copy of a contiguous 16-row A slab (16*ld halves, 16B per
// lane per op) into LDS: global_load_async_to_lds_b128 -> s_wait_asynccnt 0.
__device__ __forceinline__ void stage_a_slab(_Float16* sA,
                                             const _Float16* __restrict__ src,
                                             int ld) {
  const int total = 16 * ld;  // halves; multiple of 2048
  for (int c = threadIdx.x * 8; c < total; c += 256 * 8) {
    unsigned lds = (unsigned)(uintptr_t)(&sA[c]);   // low 32 bits = LDS offset
    const _Float16* g = src + c;
    asm volatile("global_load_async_to_lds_b128 %0, %1, off"
                 :: "v"(lds), "v"(g) : "memory");
  }
#if __has_builtin(__builtin_amdgcn_s_wait_asynccnt)
  __builtin_amdgcn_s_wait_asynccnt(0);
#else
  asm volatile("s_wait_asynccnt 0" ::: "memory");
#endif
  __syncthreads();
}

__device__ __forceinline__ float sigmoidf_(float v) {
  return 1.0f / (1.0f + __expf(-v));
}

// Device-scope grid barrier (monotonic counter, wrap-safe, 32 blocks resident).
__device__ __forceinline__ void grid_barrier(unsigned* bar, unsigned nblocks) {
  __threadfence();            // release this block's prior global writes
  __syncthreads();
  if (threadIdx.x == 0) {
    unsigned old = __hip_atomic_fetch_add(bar, 1u, __ATOMIC_RELEASE,
                                          __HIP_MEMORY_SCOPE_AGENT);
    unsigned target = (old - (old & (nblocks - 1))) + nblocks;
    while ((int)(__hip_atomic_load(bar, __ATOMIC_ACQUIRE,
                                   __HIP_MEMORY_SCOPE_AGENT) - target) < 0) {
      __builtin_amdgcn_s_sleep(2);
    }
  }
  __syncthreads();
  __threadfence();            // acquire for all threads in the block
}

// ---------- utility kernels ----------

__global__ void __launch_bounds__(256) k_cvt(const float* __restrict__ s,
                                             _Float16* __restrict__ d, int n) {
  int i = blockIdx.x * blockDim.x + threadIdx.x;
  int stride = gridDim.x * blockDim.x;
  for (; i < n; i += stride) d[i] = (_Float16)s[i];
}

__global__ void __launch_bounds__(256) k_zero_f32(float* __restrict__ d, int n) {
  int i = blockIdx.x * blockDim.x + threadIdx.x;
  if (i < n) d[i] = 0.0f;
}

__global__ void __launch_bounds__(256) k_zero_f16(_Float16* __restrict__ d, int n) {
  int i = blockIdx.x * blockDim.x + threadIdx.x;
  if (i < n) d[i] = (_Float16)0.0f;
}

__global__ void __launch_bounds__(64) k_zero_u32(unsigned* __restrict__ d, int n) {
  int i = blockIdx.x * blockDim.x + threadIdx.x;
  if (i < n) d[i] = 0u;
}

// ---------- input projection: out[s][b][h] = X16[m=b*S+s][:] . W[h][:] + bias[h] ----------
// grid: (B*S/16) * 8 blocks; block = {tm = bx>>3, tn = (bx&7)*8 + wave}.

__global__ void __launch_bounds__(256) k_xproj(const _Float16* __restrict__ X,
                                               int ldx,
                                               const _Float16* __restrict__ W,
                                               const float* __restrict__ bias,
                                               float* __restrict__ out) {
  __shared__ _Float16 sA[16 * HH];
  const int tm = blockIdx.x >> 3;
  const int tn = ((blockIdx.x & 7) << 3) + (threadIdx.x >> 5);
  stage_a_slab(sA, X + (size_t)(tm * 16) * ldx, ldx);

  const _Float16* Brow = W + (size_t)(tn * 16) * ldx;
  v8f c = {};
#pragma unroll 4
  for (int k = 0; k < ldx; k += 32) {
    v16h a = load_tile_lds(sA, ldx, k);
    v16h b = load_tile_f16(Brow + k, ldx);
    c = wmma_f16(a, b, c);
  }
  const int lane = threadIdx.x & 31;
  const int hcol = tn * 16 + (lane & 15);
  const float bv = bias[hcol];
  const int mofs = (lane >> 4) * 8;
#pragma unroll
  for (int i = 0; i < 8; ++i) {
    int m  = tm * 16 + mofs + i;
    int b_ = m >> 8;             // / S
    int s_ = m & 255;            // % S
    out[(size_t)s_ * (BB * HH) + (size_t)b_ * HH + hcol] = c[i] + bv;
  }
}

// ---------- persistent recurrent scan for one layer ----------
// 32 blocks x 256 threads; wave tile {tm = bx>>3, tn = (bx&7)*8 + wave} fixed
// for all 256 steps. Two grid barriers per step (after z/r, after g).

__global__ void __launch_bounds__(256) k_scan(const _Float16* __restrict__ Wz,
                                              const _Float16* __restrict__ Wr,
                                              const _Float16* __restrict__ Wg,
                                              const float* __restrict__ xz,
                                              const float* __restrict__ xr,
                                              const float* __restrict__ xg,
                                              float* __restrict__ h32,
                                              _Float16* __restrict__ h16,
                                              _Float16* __restrict__ y16,
                                              float* __restrict__ z,
                                              _Float16* __restrict__ rh16,
                                              unsigned* __restrict__ bar) {
  __shared__ _Float16 sA[16 * HH];
  const int tm = blockIdx.x >> 3;
  const int tn = ((blockIdx.x & 7) << 3) + (threadIdx.x >> 5);
  const int lane = threadIdx.x & 31;
  const int hcol = tn * 16 + (lane & 15);
  const int mofs = (lane >> 4) * 8;
  const _Float16* Bz = Wz + (size_t)(tn * 16) * HH;
  const _Float16* Br = Wr + (size_t)(tn * 16) * HH;
  const _Float16* Bg = Wg + (size_t)(tn * 16) * HH;

  for (int t = 0; t < SS; ++t) {
    const float* xzt = xz + (size_t)t * (BB * HH);
    const float* xrt = xr + (size_t)t * (BB * HH);
    const float* xgt = xg + (size_t)t * (BB * HH);

    // ---- phase 1: z = sig(xz + h Whz^T), r = sig(xr + h Whr^T), rh16 = f16(r*h)
    stage_a_slab(sA, h16 + (size_t)(tm * 16) * HH, HH);
    v8f cz = {}, cr = {};
#pragma unroll 4
    for (int k = 0; k < HH; k += 32) {
      v16h a  = load_tile_lds(sA, HH, k);   // shared A tile for both gates
      v16h bz = load_tile_f16(Bz + k, HH);
      cz = wmma_f16(a, bz, cz);
      v16h br = load_tile_f16(Br + k, HH);
      cr = wmma_f16(a, br, cr);
    }
#pragma unroll
    for (int i = 0; i < 8; ++i) {
      int b_ = tm * 16 + mofs + i;
      size_t idx = (size_t)b_ * HH + hcol;
      float zv = sigmoidf_(cz[i] + xzt[idx]);
      float rv = sigmoidf_(cr[i] + xrt[idx]);
      z[idx]    = zv;
      rh16[idx] = (_Float16)(rv * h32[idx]);
    }
    grid_barrier(bar, 32);

    // ---- phase 2: g = tanh(xg + rh Whg^T); h = z*h + (1-z)*g
    stage_a_slab(sA, rh16 + (size_t)(tm * 16) * HH, HH);
    v8f cg = {};
#pragma unroll 4
    for (int k = 0; k < HH; k += 32) {
      v16h a = load_tile_lds(sA, HH, k);
      v16h b = load_tile_f16(Bg + k, HH);
      cg = wmma_f16(a, b, cg);
    }
#pragma unroll
    for (int i = 0; i < 8; ++i) {
      int b_ = tm * 16 + mofs + i;
      size_t idx = (size_t)b_ * HH + hcol;
      float g  = tanhf(cg[i] + xgt[idx]);
      float zv = z[idx];
      float hn = zv * h32[idx] + (1.0f - zv) * g;
      h32[idx] = hn;
      _Float16 hh = (_Float16)hn;
      h16[idx] = hh;
      y16[(size_t)b_ * (SS * HH) + (size_t)t * HH + hcol] = hh;
    }
    grid_barrier(bar, 32);
  }
}

// ---------- output projection: out[m][o] = Y16[m][:] . Why[o][:] + bhy[o] ----------
// grid: (B*S/16) * 4 blocks; block = {tm = bx>>2, tn = (bx&3)*8 + wave}.

__global__ void __launch_bounds__(256) k_yproj(const _Float16* __restrict__ Y,
                                               const _Float16* __restrict__ W,
                                               const float* __restrict__ bias,
                                               float* __restrict__ out) {
  __shared__ _Float16 sA[16 * HH];
  const int tm = blockIdx.x >> 2;
  const int tn = ((blockIdx.x & 3) << 3) + (threadIdx.x >> 5);
  stage_a_slab(sA, Y + (size_t)(tm * 16) * HH, HH);

  const _Float16* Brow = W + (size_t)(tn * 16) * HH;
  v8f c = {};
#pragma unroll 4
  for (int k = 0; k < HH; k += 32) {
    v16h a = load_tile_lds(sA, HH, k);
    v16h b = load_tile_f16(Brow + k, HH);
    c = wmma_f16(a, b, c);
  }
  const int lane = threadIdx.x & 31;
  const int ocol = tn * 16 + (lane & 15);
  const float bv = bias[ocol];
  const int mofs = (lane >> 4) * 8;
#pragma unroll
  for (int i = 0; i < 8; ++i) {
    int m = tm * 16 + mofs + i;
    out[(size_t)m * OO + ocol] = c[i] + bv;
  }
}

// hidden_state: (B, 2, H) from final h32 of each layer
__global__ void __launch_bounds__(256) k_hcopy(const float* __restrict__ h0,
                                               const float* __restrict__ h1,
                                               float* __restrict__ out) {
  int i = blockIdx.x * blockDim.x + threadIdx.x;
  if (i < BB * HH) {
    int b  = i / HH;
    int hh = i - b * HH;
    out[(size_t)b * 2 * HH + hh]      = h0[i];
    out[(size_t)b * 2 * HH + HH + hh] = h1[i];
  }
}

// ---------- host launch ----------

extern "C" void kernel_launch(void* const* d_in, const int* in_sizes, int n_in,
                              void* d_out, int out_size, void* d_ws, size_t ws_size,
                              hipStream_t stream) {
  (void)in_sizes; (void)n_in; (void)out_size; (void)ws_size;

  // f16 weight slots: Wxz0 Wxr0 Wxg0 | Whz0 Whr0 Whg0 | Wxz1 Wxr1 Wxg1 | Whz1 Whr1 Whg1 | Why
  const int    wsel[13] = {1, 4, 7,  3, 6, 9,  10, 13, 16,  12, 15, 18,  19};
  const size_t wcnt[13] = {(size_t)HH*II, (size_t)HH*II, (size_t)HH*II,
                           (size_t)HH*HH, (size_t)HH*HH, (size_t)HH*HH,
                           (size_t)HH*HH, (size_t)HH*HH, (size_t)HH*HH,
                           (size_t)HH*HH, (size_t)HH*HH, (size_t)HH*HH,
                           (size_t)OO*HH};

  char* ws = (char*)d_ws;
  size_t off = 0;
  auto take = [&](size_t bytes) -> void* {
    void* p = ws + off;
    off = (off + bytes + 255) & ~(size_t)255;
    return p;
  };

  _Float16* wf[13];
  for (int i = 0; i < 13; ++i) wf[i] = (_Float16*)take(wcnt[i] * sizeof(_Float16));
  _Float16* x16 = (_Float16*)take((size_t)BB * SS * II * sizeof(_Float16));
  _Float16* y16 = (_Float16*)take((size_t)BB * SS * HH * sizeof(_Float16));
  float* xz = (float*)take((size_t)SS * BB * HH * sizeof(float));
  float* xr = (float*)take((size_t)SS * BB * HH * sizeof(float));
  float* xg = (float*)take((size_t)SS * BB * HH * sizeof(float));
  float*    h32_0 = (float*)take((size_t)BB * HH * sizeof(float));
  float*    h32_1 = (float*)take((size_t)BB * HH * sizeof(float));
  _Float16* h16_0 = (_Float16*)take((size_t)BB * HH * sizeof(_Float16));
  _Float16* h16_1 = (_Float16*)take((size_t)BB * HH * sizeof(_Float16));
  float*    zb    = (float*)take((size_t)BB * HH * sizeof(float));
  _Float16* rh16  = (_Float16*)take((size_t)BB * HH * sizeof(_Float16));
  unsigned* bar   = (unsigned*)take(256);

  // 1) convert weights + x to f16 (weights then stay L2-resident: ~22 MB << 192 MB)
  for (int i = 0; i < 13; ++i) {
    int n = (int)wcnt[i];
    k_cvt<<<(n + 255) / 256, 256, 0, stream>>>((const float*)d_in[wsel[i]], wf[i], n);
  }
  {
    int n = BB * SS * II;
    k_cvt<<<(n + 255) / 256, 256, 0, stream>>>((const float*)d_in[0], x16, n);
  }
  k_zero_u32<<<1, 64, 0, stream>>>(bar, 1);

  const int xproj_grid = (BB * SS / 16) * 8;  // 8 N-tiles per block (8 waves)
  const int scan_grid  = (BB / 16) * 8;       // 32 blocks
  const int nzero      = BB * HH;

  // 2) layer 0
  {
    const float* bias[3] = {(const float*)d_in[2], (const float*)d_in[5], (const float*)d_in[8]};
    float* outs[3] = {xz, xr, xg};
    for (int g = 0; g < 3; ++g)
      k_xproj<<<xproj_grid, 256, 0, stream>>>(x16, II, wf[g], bias[g], outs[g]);
  }
  k_zero_f32<<<(nzero + 255) / 256, 256, 0, stream>>>(h32_0, nzero);
  k_zero_f16<<<(nzero + 255) / 256, 256, 0, stream>>>(h16_0, nzero);
  k_scan<<<scan_grid, 256, 0, stream>>>(wf[3], wf[4], wf[5], xz, xr, xg,
                                        h32_0, h16_0, y16, zb, rh16, bar);

  // 3) layer 1 (input projections consume y16 from layer 0, then scan overwrites y16)
  {
    const float* bias[3] = {(const float*)d_in[11], (const float*)d_in[14], (const float*)d_in[17]};
    float* outs[3] = {xz, xr, xg};
    for (int g = 0; g < 3; ++g)
      k_xproj<<<xproj_grid, 256, 0, stream>>>(y16, HH, wf[6 + g], bias[g], outs[g]);
  }
  k_zero_f32<<<(nzero + 255) / 256, 256, 0, stream>>>(h32_1, nzero);
  k_zero_f16<<<(nzero + 255) / 256, 256, 0, stream>>>(h16_1, nzero);
  k_scan<<<scan_grid, 256, 0, stream>>>(wf[9], wf[10], wf[11], xz, xr, xg,
                                        h32_1, h16_1, y16, zb, rh16, bar);

  // 4) output projection + hidden states
  k_yproj<<<(BB * SS / 16) * 4, 256, 0, stream>>>(y16, wf[12], (const float*)d_in[20],
                                                  (float*)d_out);
  k_hcopy<<<(nzero + 255) / 256, 256, 0, stream>>>(h32_0, h32_1,
                                                   (float*)d_out + (size_t)BB * SS * OO);
}
